// GAT_1726576854973
// MI455X (gfx1250) — matrix operand order repaired and checked by reading, not compiled, not run on hardware
//
#include <hip/hip_runtime.h>

#define NEG_SLOPE 0.2f

typedef __attribute__((ext_vector_type(16))) __bf16 v16bf;
typedef __attribute__((ext_vector_type(8)))  float  v8f;

// ---------------- helpers ----------------

__device__ __forceinline__ unsigned short f2bf(float f) {
  unsigned u = __float_as_uint(f);
  unsigned r = u + 0x7FFFu + ((u >> 16) & 1u);   // round-to-nearest-even
  return (unsigned short)(r >> 16);
}

__device__ __forceinline__ unsigned enc_f(float f) {  // order-preserving float->uint
  unsigned u = __float_as_uint(f);
  return (u & 0x80000000u) ? ~u : (u | 0x80000000u);
}
__device__ __forceinline__ float dec_f(unsigned u) {
  return __uint_as_float((u & 0x80000000u) ? (u ^ 0x80000000u) : ~u);
}

__device__ __forceinline__ void edge_sd(const int* __restrict__ ei, int e, int E,
                                        int& src, int& dst) {
  if (e < E) { src = ei[e]; dst = ei[E + e]; }
  else       { src = e - E; dst = e - E; }        // implicit self-loops
}

union frag_u { unsigned u[8]; v16bf v; };

// ---------------- elementwise / pack kernels ----------------

__global__ void k_f32_to_bf16(const float* __restrict__ src,
                              unsigned short* __restrict__ dst, long long n) {
  long long i = (long long)blockIdx.x * blockDim.x + threadIdx.x;
  if (i < n) dst[i] = f2bf(src[i]);
}

// Pack fp32 weight [K, Nout] row-major into WMMA B-fragment tiles of 256 dwords.
// Tile t = tn*kTiles + kt ; element r = v*32 + lane ;
// lane: n = tn*16 + (lane&15), half = lane>>4 ; k = kt*32 + half*16 + 2v
__global__ void k_pack_b(const float* __restrict__ W, unsigned* __restrict__ P,
                         int K, int Nout) {
  int kTiles = K >> 5;
  long long total = (long long)(Nout >> 4) * kTiles * 256;
  long long t = (long long)blockIdx.x * blockDim.x + threadIdx.x;
  if (t >= total) return;
  int r = (int)(t & 255);
  long long tile = t >> 8;
  int v = r >> 5, lane = r & 31;
  int tn = (int)(tile / kTiles), kt = (int)(tile % kTiles);
  int n = tn * 16 + (lane & 15);
  int k = kt * 32 + (lane >> 4) * 16 + v * 2;
  unsigned lo = f2bf(W[(size_t)k * Nout + n]);
  unsigned hi = f2bf(W[(size_t)(k + 1) * Nout + n]);
  P[t] = lo | (hi << 16);
}

// ---------------- WMMA GEMM with 2x2 register blocking ----------------
// One wave computes a 32x32 output block (4 x 16x16 WMMA accumulators).
// Per K-step: 2 A-frags + 2 B-frags -> 4 x v_wmma_f32_16x16x32_bf16.
// Requires N % 32 == 0 (N in {1024, 384, 64}); M tiles clamped on the odd tail.
__global__ void k_gemm(const unsigned short* __restrict__ A,
                       const unsigned* __restrict__ Bp,
                       float* __restrict__ C,
                       const float* __restrict__ bias,
                       int M, int N, int K) {
  int wave = (int)(((long long)blockIdx.x * blockDim.x + threadIdx.x) >> 5);
  int lane = threadIdx.x & 31;
  int nT2 = N >> 5;                       // N / 32
  int kTiles = K >> 5, mTiles = M >> 4;
  int mT2 = (mTiles + 1) >> 1;
  if (wave >= mT2 * nT2) return;          // wave-uniform: EXEC stays all-1s
  int pm = wave / nT2, pn = wave - pm * nT2;
  int tm0 = pm * 2;
  bool hasM1 = (tm0 + 1) < mTiles;        // uniform per wave
  int tm1 = hasM1 ? tm0 + 1 : tm0;        // clamp loads on the odd tail
  int tn0 = pn * 2, tn1 = tn0 + 1;
  int half = lane >> 4, nl = lane & 15;

  const unsigned* Ar0 = (const unsigned*)(A + (size_t)(tm0 * 16 + nl) * K);  // K even
  const unsigned* Ar1 = (const unsigned*)(A + (size_t)(tm1 * 16 + nl) * K);
  const unsigned* Bt0 = Bp + (size_t)tn0 * kTiles * 256 + lane;
  const unsigned* Bt1 = Bp + (size_t)tn1 * kTiles * 256 + lane;

  v8f acc00 = {0.f,0.f,0.f,0.f,0.f,0.f,0.f,0.f};
  v8f acc01 = acc00, acc10 = acc00, acc11 = acc00;

  for (int kt = 0; kt < kTiles; ++kt) {
    frag_u a0, a1, b0, b1;
    int pb = kt * 16;                     // bf16-pair index base for this k-block
#pragma unroll
    for (int q = 0; q < 8; ++q) {         // A frag: pair = half*8 + q
      int off = pb + half * 8 + q;
      a0.u[q] = Ar0[off];
      a1.u[q] = Ar1[off];
    }
#pragma unroll
    for (int q = 0; q < 8; ++q) {         // B frags: pre-packed, coalesced
      b0.u[q] = Bt0[kt * 256 + q * 32];
      b1.u[q] = Bt1[kt * 256 + q * 32];
    }
    acc00 = __builtin_amdgcn_wmma_f32_16x16x32_bf16(false, a0.v, false, b0.v,
                                                    (short)0, acc00, false, false);
    acc01 = __builtin_amdgcn_wmma_f32_16x16x32_bf16(false, a0.v, false, b1.v,
                                                    (short)0, acc01, false, false);
    acc10 = __builtin_amdgcn_wmma_f32_16x16x32_bf16(false, a1.v, false, b0.v,
                                                    (short)0, acc10, false, false);
    acc11 = __builtin_amdgcn_wmma_f32_16x16x32_bf16(false, a1.v, false, b1.v,
                                                    (short)0, acc11, false, false);
  }

  int n0 = tn0 * 16 + nl, n1 = tn1 * 16 + nl;
  float bv0 = bias ? bias[n0] : 0.f;
  float bv1 = bias ? bias[n1] : 0.f;
#pragma unroll
  for (int q = 0; q < 8; ++q) {           // C/D layout: lanes 16-31 hold M=8..15
    size_t row0 = (size_t)(tm0 * 16 + half * 8 + q);
    C[row0 * N + n0] = acc00[q] + bv0;
    C[row0 * N + n1] = acc01[q] + bv1;
  }
  if (hasM1) {
#pragma unroll
    for (int q = 0; q < 8; ++q) {
      size_t row1 = (size_t)(tm1 * 16 + half * 8 + q);
      C[row1 * N + n0] = acc10[q] + bv0;
      C[row1 * N + n1] = acc11[q] + bv1;
    }
  }
}

// ---------------- attention kernels ----------------

// s[n,h] = sum_c h[n,h,c]*a_src[h,c] ; d likewise. Node arrays use stride 8 over heads.
__global__ void k_logits(const float* __restrict__ h,
                         const float* __restrict__ a_src,
                         const float* __restrict__ a_dst,
                         float* __restrict__ s, float* __restrict__ d,
                         int N, int H, int C) {
  int i = blockIdx.x * blockDim.x + threadIdx.x;
  if (i >= N * H) return;
  int n = i / H, hh = i - n * H;
  const float* hp = h + (size_t)n * H * C + (size_t)hh * C;
  const float* as = a_src + hh * C;
  const float* ad = a_dst + hh * C;
  float ss = 0.f, dd = 0.f;
  for (int c = 0; c < C; ++c) { float v = hp[c]; ss += v * as[c]; dd += v * ad[c]; }
  s[n * 8 + hh] = ss;
  d[n * 8 + hh] = dd;
}

__global__ void k_edge_max(const int* __restrict__ ei, int E, int N, int H,
                           const float* __restrict__ s, const float* __restrict__ d,
                           unsigned* __restrict__ menc) {
  long long i = (long long)blockIdx.x * blockDim.x + threadIdx.x;
  long long tot = (long long)(E + N) * H;
  if (i >= tot) return;
  int e = (int)(i / H), hh = (int)(i % H);
  int src, dst; edge_sd(ei, e, E, src, dst);
  float v = s[src * 8 + hh] + d[dst * 8 + hh];
  v = v > 0.f ? v : NEG_SLOPE * v;
  atomicMax(&menc[dst * 8 + hh], enc_f(v));
}

__global__ void k_edge_exp(const int* __restrict__ ei, int E, int N, int H,
                           const float* __restrict__ s, const float* __restrict__ d,
                           const unsigned* __restrict__ menc,
                           float* __restrict__ den, float* __restrict__ ew) {
  long long i = (long long)blockIdx.x * blockDim.x + threadIdx.x;
  long long tot = (long long)(E + N) * H;
  if (i >= tot) return;
  int e = (int)(i / H), hh = (int)(i % H);
  int src, dst; edge_sd(ei, e, E, src, dst);
  float v = s[src * 8 + hh] + d[dst * 8 + hh];
  v = v > 0.f ? v : NEG_SLOPE * v;
  float m = dec_f(menc[dst * 8 + hh]);
  float w = expf(v - m);
  ew[(size_t)e * H + hh] = w;
  atomicAdd(&den[dst * 8 + hh], w);
}

// out[dst, h*C+c] += h[src, h*C+c] * alpha(e,h)
__global__ void k_scatter(const int* __restrict__ ei, int E, int N, int H, int C,
                          const float* __restrict__ hmat,
                          const float* __restrict__ ew,
                          const float* __restrict__ den,
                          float* __restrict__ out) {
  long long i = (long long)blockIdx.x * blockDim.x + threadIdx.x;
  long long HC = (long long)H * C;
  long long tot = (long long)(E + N) * HC;
  if (i >= tot) return;
  int e = (int)(i / HC);
  int f = (int)(i - (long long)e * HC);
  int hh = f / C;
  int src, dst; edge_sd(ei, e, E, src, dst);
  float alpha = ew[(size_t)e * H + hh] / den[dst * 8 + hh];
  atomicAdd(&out[(size_t)dst * HC + f], hmat[(size_t)src * HC + f] * alpha);
}

// ---------------- combine kernels ----------------

__global__ void k_combine_concat(const float* __restrict__ go,
                                 const float* __restrict__ lo,
                                 const float* __restrict__ b,
                                 unsigned short* __restrict__ actout,
                                 long long total, int F) {
  long long i = (long long)blockIdx.x * blockDim.x + threadIdx.x;
  if (i >= total) return;
  int f = (int)(i % F);
  float v = go[i] + b[f] + lo[i];
  v = v > 0.f ? v : (expf(v) - 1.f);              // ELU
  actout[i] = f2bf(v);
}

__global__ void k_combine_mean(const float* __restrict__ go,
                               const float* __restrict__ lo,
                               const float* __restrict__ b,
                               float* __restrict__ out, int N, int H, int C) {
  int i = blockIdx.x * blockDim.x + threadIdx.x;
  if (i >= N * C) return;
  int n = i / C, c = i - n * C;
  float acc = 0.f;
  for (int hh = 0; hh < H; ++hh) acc += go[(size_t)n * H * C + (size_t)hh * C + c];
  out[i] = acc / (float)H + b[c] + lo[i];
}

// ---------------- host orchestration ----------------

extern "C" void kernel_launch(void* const* d_in, const int* in_sizes, int n_in,
                              void* d_out, int out_size, void* d_ws, size_t ws_size,
                              hipStream_t stream) {
  (void)n_in; (void)out_size; (void)ws_size;
  const float* x   = (const float*)d_in[0];
  const int*   ei  = (const int*)d_in[1];
  const float* W1  = (const float*)d_in[2];
  const float* a1s = (const float*)d_in[3];
  const float* a1d = (const float*)d_in[4];
  const float* b1  = (const float*)d_in[5];
  const float* l1W = (const float*)d_in[6];
  const float* l1b = (const float*)d_in[7];
  const float* W2  = (const float*)d_in[8];
  const float* a2s = (const float*)d_in[9];
  const float* a2d = (const float*)d_in[10];
  const float* b2  = (const float*)d_in[11];
  const float* l2W = (const float*)d_in[12];
  const float* l2b = (const float*)d_in[13];
  const float* W3  = (const float*)d_in[14];
  const float* a3s = (const float*)d_in[15];
  const float* a3d = (const float*)d_in[16];
  const float* b3  = (const float*)d_in[17];
  const float* l3W = (const float*)d_in[18];
  const float* l3b = (const float*)d_in[19];

  const int IN = 128, HID = 1024;
  const int N = in_sizes[0] / IN;       // 10000
  const int E = in_sizes[1] / 2;        // 160000

  char* p = (char*)d_ws;
  auto carve = [&](size_t bytes) -> char* {
    char* r = p; p += (bytes + 255) & ~(size_t)255; return r;
  };
  unsigned short* act  = (unsigned short*)carve((size_t)N * HID * 2); // activations (bf16)
  unsigned* pack1      = (unsigned*)carve((size_t)2 << 20);           // packed W
  unsigned* pack2      = (unsigned*)carve((size_t)2 << 20);           // packed lin_W
  float* gath          = (float*)carve((size_t)N * HID * 4);          // h = act @ W
  float* lino          = (float*)carve((size_t)N * HID * 4);          // act @ lin_W + lin_b
  float* gato          = (float*)carve((size_t)N * HID * 4);          // aggregated messages
  float* slog          = (float*)carve((size_t)N * 8 * 4);
  float* dlog          = (float*)carve((size_t)N * 8 * 4);
  unsigned* menc       = (unsigned*)carve((size_t)N * 8 * 4);
  float* den           = (float*)carve((size_t)N * 8 * 4);
  float* ew            = (float*)carve((size_t)(E + N) * 8 * 4);

  const int T = 256;
  auto cd = [](long long a, long long b) { return (int)((a + b - 1) / b); };

  // x -> bf16
  k_f32_to_bf16<<<cd((long long)N * IN, T), T, 0, stream>>>(x, act, (long long)N * IN);

  auto gemm = [&](const unsigned short* A, const float* W, unsigned* pk,
                  float* Cm, const float* bias, int M, int Nout, int K) {
    long long pkt = (long long)(Nout >> 4) * (K >> 5) * 256;
    k_pack_b<<<cd(pkt, T), T, 0, stream>>>(W, pk, K, Nout);
    int mT2 = ((M >> 4) + 1) >> 1;
    long long waves = (long long)mT2 * (Nout >> 5);
    k_gemm<<<cd(waves * 32, T), T, 0, stream>>>(A, pk, Cm, bias, M, Nout, K);
  };

  auto gat_edges = [&](int H, int C) {
    hipMemsetAsync(menc, 0, (size_t)N * 8 * 4, stream);   // enc(-inf) < 0x01000000
    hipMemsetAsync(den, 0, (size_t)N * 8 * 4, stream);
    hipMemsetAsync(gato, 0, (size_t)N * H * C * 4, stream);
    long long eh = (long long)(E + N) * H;
    k_edge_max<<<cd(eh, T), T, 0, stream>>>(ei, E, N, H, slog, dlog, menc);
    k_edge_exp<<<cd(eh, T), T, 0, stream>>>(ei, E, N, H, slog, dlog, menc, den, ew);
    long long tot = (long long)(E + N) * H * C;
    k_scatter<<<cd(tot, T), T, 0, stream>>>(ei, E, N, H, C, gath, ew, den, gato);
  };

  // ---- layer 1: GATConv(128->256, H=4) + lin(128->1024), ELU ----
  gemm(act, W1, pack1, gath, nullptr, N, HID, IN);
  gemm(act, l1W, pack2, lino, l1b, N, HID, IN);
  k_logits<<<cd((long long)N * 4, T), T, 0, stream>>>(gath, a1s, a1d, slog, dlog, N, 4, 256);
  gat_edges(4, 256);
  k_combine_concat<<<cd((long long)N * HID, T), T, 0, stream>>>(gato, lino, b1, act,
                                                                (long long)N * HID, HID);

  // ---- layer 2: GATConv(1024->256, H=4) + lin(1024->1024), ELU ----
  gemm(act, W2, pack1, gath, nullptr, N, HID, HID);
  gemm(act, l2W, pack2, lino, l2b, N, HID, HID);
  k_logits<<<cd((long long)N * 4, T), T, 0, stream>>>(gath, a2s, a2d, slog, dlog, N, 4, 256);
  gat_edges(4, 256);
  k_combine_concat<<<cd((long long)N * HID, T), T, 0, stream>>>(gato, lino, b2, act,
                                                                (long long)N * HID, HID);

  // ---- layer 3: GATConv(1024->64, H=6, mean) + lin(1024->64) ----
  gemm(act, W3, pack1, gath, nullptr, N, 384, HID);
  gemm(act, l3W, pack2, lino, l3b, N, 64, HID);
  k_logits<<<cd((long long)N * 6, T), T, 0, stream>>>(gath, a3s, a3d, slog, dlog, N, 6, 64);
  gat_edges(6, 64);
  k_combine_mean<<<cd((long long)N * 64, T), T, 0, stream>>>(gato, lino, b3,
                                                             (float*)d_out, N, 6, 64);
}
